// SceneDecoder_53395033424423
// MI455X (gfx1250) — compile-verified, matrix operand order checked by reading, not run on hardware
//
#include <hip/hip_runtime.h>

// ---------------------------------------------------------------------------
// SceneDecoder for MI455X (gfx1250), wave32 + WMMA + async-to-LDS staging.
//
// Algebraic refactoring (exact up to f16 GEMM rounding):
//   * segment_sum(x[src]@W, dst)  ->  segment_sum(x[src], dst) @ W
//   * concat(x[src],x[dst])@W1    ->  (x@W1_top)[src] + (x@W1_bot)[dst]
// leaving ~15 GFLOP of dense GEMM (v_wmma_f32_16x16x32_f16) and L2-resident
// gather/scatter streams (all tables are 4MB << 192MB L2).
// Activations keep f16 mirrors so the GEMM hot loop has no cvt instructions;
// B panels are staged to LDS with global_load_async_to_lds_b128 and shared by
// all 8 waves of a block.
// ---------------------------------------------------------------------------

typedef __attribute__((ext_vector_type(16))) _Float16 v16h;
typedef __attribute__((ext_vector_type(8)))  _Float16 v8h;
typedef __attribute__((ext_vector_type(8)))  float    v8f;
typedef __attribute__((ext_vector_type(4)))  float    v4f;

static __device__ __forceinline__ float wred(float v) {
#pragma unroll
  for (int o = 16; o > 0; o >>= 1) v += __shfl_xor(v, o, 32);
  return v;
}

// --------------------------- weight repack (f32 -> f16 fragment order) -----
// Packed index: ((nt*(K/32)+kt)*32 + lane)*16 + j
// element (lane,j) holds B[k][n] with n = nt*16 + lane%16,
//   k = kt*32 + (j%8) + (j/8)*16 + (lane/16)*8   (mirrors documented A layout)
__global__ __launch_bounds__(256) void pack_b_kernel(
    const float* __restrict__ B, _Float16* __restrict__ Bp, int K, int N) {
  int idx = blockIdx.x * 256 + threadIdx.x;
  if (idx >= K * N) return;
  int j    = idx & 15;
  int lane = (idx >> 4) & 31;
  int t    = idx >> 9;                 // nt*(K/32)+kt
  int KT   = K >> 5;
  int nt   = t / KT;
  int kt   = t - nt * KT;
  int n = nt * 16 + (lane & 15);
  int k = kt * 32 + (j & 7) + ((j >> 3) << 4) + ((lane >> 4) << 3);
  Bp[idx] = (_Float16)B[(size_t)k * N + n];
}

__global__ __launch_bounds__(256) void cvt_f16_kernel(
    const float* __restrict__ s, _Float16* __restrict__ d, int n) {
  int i = blockIdx.x * 256 + threadIdx.x;
  if (i < n) d[i] = (_Float16)s[i];
}

// --------------------------- WMMA GEMM -------------------------------------
// C = act(A[M,K] @ B + bias); A: f16 row-major; B: packed f16 (LDS-staged);
// outputs: optional f32 C32 and/or f16 C16 (row-major, stride N).
// Block = 8 waves covering 8 M-tiles x one 32-wide N strip; the strip's B
// panel (contiguous in the packed layout) is async-copied to LDS once.
__global__ __launch_bounds__(256) void gemm_wmma_kernel(
    const _Float16* __restrict__ A, const _Float16* __restrict__ Bp,
    const float* __restrict__ bias, float* __restrict__ C32,
    _Float16* __restrict__ C16, int M, int K, int N, int relu) {
  __shared__ _Float16 sB[2 * 16 * 512];           // up to K=512: 32KB panel
  const int wave = threadIdx.x >> 5;
  const int lane = threadIdx.x & 31;
  const int KT   = K >> 5;
  const int NT2  = N >> 5;
  const int MT   = M >> 4;
  const int nt2  = blockIdx.x % NT2;
  const int mb   = blockIdx.x / NT2;
  const int nt0  = nt2 * 2;
  const int mt   = mb * 8 + wave;                 // wave-uniform

  // ---- stage the 32-column B panel (2*KT fragments, contiguous) to LDS ----
  {
    const uint64_t gbase = (uint64_t)(Bp + (size_t)nt0 * KT * 512);
    const uint32_t lbase = (uint32_t)(uintptr_t)(&sB[0]);
    const int nchunks = KT * 128;                 // 16B chunks, mult. of 256
    for (int c = threadIdx.x; c < nchunks; c += 256) {
      uint32_t ldst = lbase + c * 16;
      uint64_t gsrc = gbase + (uint64_t)c * 16;
      asm volatile("global_load_async_to_lds_b128 %0, %1, off"
                   :: "v"(ldst), "v"(gsrc) : "memory");
    }
    asm volatile("s_wait_asynccnt 0x0" ::: "memory");
  }
  __syncthreads();

  if (mt < MT) {                                  // wave-uniform guard
    const int r  = lane & 15;
    const int hi = lane >> 4;
    const _Float16* Arow = A + (size_t)(mt * 16 + r) * K + hi * 8;

    v8f acc[2] = {{}, {}};
    for (int kt = 0; kt < KT; ++kt) {
      v8h a0 = *(const v8h*)(Arow + kt * 32);
      v8h a1 = *(const v8h*)(Arow + kt * 32 + 16);
      v16h av = __builtin_shufflevector(a0, a1, 0, 1, 2, 3, 4, 5, 6, 7,
                                        8, 9, 10, 11, 12, 13, 14, 15);
      v16h bv0 = *(const v16h*)(sB + ((size_t)kt) * 512 + lane * 16);
      v16h bv1 = *(const v16h*)(sB + ((size_t)(KT + kt)) * 512 + lane * 16);
      acc[0] = __builtin_amdgcn_wmma_f32_16x16x32_f16(false, av, false, bv0,
                                                      (short)0, acc[0], false, false);
      acc[1] = __builtin_amdgcn_wmma_f32_16x16x32_f16(false, av, false, bv1,
                                                      (short)0, acc[1], false, false);
    }
#pragma unroll
    for (int t = 0; t < 2; ++t) {
      int   n  = (nt0 + t) * 16 + r;
      float bb = bias ? bias[n] : 0.f;
#pragma unroll
      for (int j = 0; j < 8; ++j) {
        int   m = mt * 16 + j + hi * 8;   // C layout: vgpr j -> row j (+8 hi)
        float v = acc[t][j] + bb;
        if (relu) v = fmaxf(v, 0.f);
        if (C32) C32[(size_t)m * N + n] = v;
        if (C16) C16[(size_t)m * N + n] = (_Float16)v;
      }
    }
  }
}

// --------------------------- misc elementwise kernels ----------------------
__global__ __launch_bounds__(256) void fill_zero_kernel(float* p, int n) {
  int i = blockIdx.x * 256 + threadIdx.x;
  if (i < n) p[i] = 0.f;
}

// Acat16[r, 0:256] = node_queries[r%64], Acat16[r, 256:512] = zh[r/64]
__global__ __launch_bounds__(256) void concat_qz_kernel(
    _Float16* __restrict__ o, const float* __restrict__ q, const float* __restrict__ zh) {
  int idx = blockIdx.x * 256 + threadIdx.x;
  if (idx >= 4096 * 512) return;
  int r = idx >> 9, c = idx & 511;
  float v = (c < 256) ? q[(r & 63) * 256 + c] : zh[(r >> 6) * 256 + (c - 256)];
  o[idx] = (_Float16)v;
}

// Acat16[r, 0:256] = x[r], Acat16[r, 256:512] = zh[r/64]
__global__ __launch_bounds__(256) void concat_xz_kernel(
    _Float16* __restrict__ o, const float* __restrict__ x, const float* __restrict__ zh) {
  int idx = blockIdx.x * 256 + threadIdx.x;
  if (idx >= 4096 * 512) return;
  int r = idx >> 9, c = idx & 511;
  float v = (c < 256) ? x[(size_t)r * 256 + c] : zh[(r >> 6) * 256 + (c - 256)];
  o[idx] = (_Float16)v;
}

// g[dst] += x[src], one wave per edge; atomics land in the 4MB L2-resident g.
__global__ __launch_bounds__(256) void scatter_kernel(
    float* __restrict__ g, const float* __restrict__ x,
    const int* __restrict__ src, const int* __restrict__ dst, int E) {
  int wid  = (int)((blockIdx.x * 256 + threadIdx.x) >> 5);
  int lane = threadIdx.x & 31;
  if (wid >= E) return;
  int s = src[wid], d = dst[wid];
  const float* xs = x + (size_t)s * 256 + lane * 8;
  float*       gd = g + (size_t)d * 256 + lane * 8;
  v4f a = *(const v4f*)xs;
  v4f b = *(const v4f*)(xs + 4);
#pragma unroll
  for (int j = 0; j < 4; ++j) { atomicAdd(gd + j, a[j]); atomicAdd(gd + 4 + j, b[j]); }
}

// x = layernorm(x + u) * gam + bet (also refresh x16 mirror), wave per row.
__global__ __launch_bounds__(256) void resln_kernel(
    float* __restrict__ x, _Float16* __restrict__ x16, const float* __restrict__ u,
    const float* __restrict__ gam, const float* __restrict__ bet, int M) {
  int wid  = (int)((blockIdx.x * 256 + threadIdx.x) >> 5);
  int lane = threadIdx.x & 31;
  if (wid >= M) return;
  float*     xr = x   + (size_t)wid * 256;
  _Float16*  hr = x16 + (size_t)wid * 256;
  const float* ur = u + (size_t)wid * 256;
  int c0 = lane * 8;
  float v[8], s = 0.f;
#pragma unroll
  for (int j = 0; j < 8; ++j) { v[j] = xr[c0 + j] + ur[c0 + j]; s += v[j]; }
  float mu = wred(s) * (1.f / 256.f);
  float vs = 0.f;
#pragma unroll
  for (int j = 0; j < 8; ++j) { float d = v[j] - mu; vs += d * d; }
  float rs = rsqrtf(wred(vs) * (1.f / 256.f) + 1e-5f);
#pragma unroll
  for (int j = 0; j < 8; ++j) {
    float o = (v[j] - mu) * rs * gam[c0 + j] + bet[c0 + j];
    xr[c0 + j] = o;
    hr[c0 + j] = (_Float16)o;
  }
}

// tiny-N dense layer (N<=4): one thread per output element, f16 activations
__global__ __launch_bounds__(256) void head_small_kernel(
    const _Float16* __restrict__ A, const float* __restrict__ W,
    const float* __restrict__ b, float* __restrict__ C,
    int M, int K, int N, int relu) {
  int idx = blockIdx.x * 256 + threadIdx.x;
  if (idx >= M * N) return;
  int m = idx / N, n = idx - m * N;
  float s = b[n];
  const _Float16* a = A + (size_t)m * K;
  for (int k = 0; k < K; ++k) s += (float)a[k] * W[k * N + n];
  if (relu) s = fmaxf(s, 0.f);
  C[idx] = s;
}

__global__ __launch_bounds__(256) void pool_kernel(
    const float* __restrict__ x, float* __restrict__ pooled) {
  int idx = blockIdx.x * 256 + threadIdx.x;
  if (idx >= 64 * 256) return;
  int b = idx >> 8, c = idx & 255;
  float s = 0.f;
  for (int n = 0; n < 64; ++n) s += x[(size_t)((b << 6) + n) * 256 + c];
  pooled[idx] = s * (1.f / 64.f);
}

__global__ __launch_bounds__(256) void freq1_kernel(
    const float* __restrict__ p, const float* __restrict__ W,
    const float* __restrict__ b, float* __restrict__ h) {
  int idx = blockIdx.x * 256 + threadIdx.x;
  if (idx >= 64 * 64) return;
  int bb = idx >> 6, j = idx & 63;
  float s = b[j];
  for (int k = 0; k < 256; ++k) s += p[bb * 256 + k] * W[k * 64 + j];
  h[idx] = fmaxf(s, 0.f);
}

__global__ __launch_bounds__(64) void freq2_kernel(
    const float* __restrict__ h, const float* __restrict__ W,
    const float* __restrict__ b, float* __restrict__ o) {
  int i = blockIdx.x * 64 + threadIdx.x;
  if (i >= 64) return;
  float s = b[0];
  for (int k = 0; k < 64; ++k) s += h[i * 64 + k] * W[k];
  o[i] = s;
}

// edge heads: edist/edir from factored node tables, one wave per edge.
__global__ __launch_bounds__(256) void edge_head_kernel(
    const float* __restrict__ Pd, const float* __restrict__ Qd,
    const float* __restrict__ Pr, const float* __restrict__ Qr,
    const float* __restrict__ b1d, const float* __restrict__ w2d, const float* __restrict__ b2d,
    const float* __restrict__ b1r, const float* __restrict__ W2r, const float* __restrict__ b2r,
    const int* __restrict__ src, const int* __restrict__ dst,
    float* __restrict__ edist, float* __restrict__ edir, int E) {
  int wid  = (int)((blockIdx.x * 256 + threadIdx.x) >> 5);
  int lane = threadIdx.x & 31;
  if (wid >= E) return;
  int s = src[wid], d = dst[wid];
  int c0 = lane * 8;
  const float* pd = Pd + (size_t)s * 256 + c0;
  const float* qd = Qd + (size_t)d * 256 + c0;
  const float* pr = Pr + (size_t)s * 256 + c0;
  const float* qr = Qr + (size_t)d * 256 + c0;
  float sd = 0.f, s0 = 0.f, s1 = 0.f, s2 = 0.f;
#pragma unroll
  for (int j = 0; j < 8; ++j) {
    int   c = c0 + j;
    float h = fmaxf(pd[j] + qd[j] + b1d[c], 0.f);
    sd += h * w2d[c];
    float g = fmaxf(pr[j] + qr[j] + b1r[c], 0.f);
    s0 += g * W2r[c * 3 + 0];
    s1 += g * W2r[c * 3 + 1];
    s2 += g * W2r[c * 3 + 2];
  }
  sd = wred(sd); s0 = wred(s0); s1 = wred(s1); s2 = wred(s2);
  if (lane == 0) {
    edist[wid] = sd + b2d[0];
    edir[(size_t)wid * 3 + 0] = s0 + b2r[0];
    edir[(size_t)wid * 3 + 1] = s1 + b2r[1];
    edir[(size_t)wid * 3 + 2] = s2 + b2r[2];
  }
}

// ---------------------------------------------------------------------------
extern "C" void kernel_launch(void* const* d_in, const int* in_sizes, int n_in,
                              void* d_out, int out_size, void* d_ws, size_t ws_size,
                              hipStream_t stream) {
  const int NTOT = 4096;
  const int E = in_sizes[1] / 2;

  const float* z  = (const float*)d_in[0];
  const int*   ei = (const int*)d_in[1];
  const float* nq = (const float*)d_in[2];
  const int* src = ei;
  const int* dst = ei + E;
  auto F = [&](int i) { return (const float*)d_in[i]; };

  // input leaf order (insertion order of setup_inputs dict / nested lists):
  // 3..6: z_proj (W,b)x2 ; 7..10: combine ; 11+7l: Wmsg,uW1,ub1,uW2,ub2,g,b
  // 39..44 centroid_obj ; 45..50 centroid_txrx ; 51..54 node_type ;
  // 55..58 freq ; 59..62 edge_dist ; 63..66 edge_dir
  float* out = (float*)d_out;
  const int O_CO = 0, O_CT = 12288, O_NT = 24576, O_FREQ = 36864,
            O_ED = 36928, O_ER = 299072;

  // ---- workspace bump allocator (256B aligned) ----
  char* base = (char*)d_ws;
  size_t off = 0;
  auto alloc = [&](size_t bytes) -> void* {
    void* p = base + off;
    off += (bytes + 255) & ~(size_t)255;
    return p;
  };
  auto allocF = [&](size_t n) { return (float*)alloc(n * 4); };
  auto allocH = [&](size_t n) { return (_Float16*)alloc(n * 2); };

  _Float16* pz1 = allocH(128 * 512);
  _Float16* pz2 = allocH(512 * 256);
  _Float16* pc1 = allocH(512 * 256);
  _Float16* pc2 = allocH(256 * 256);
  _Float16* pmsg[4]; _Float16* pu1[4]; _Float16* pu2[4];
  for (int l = 0; l < 4; ++l) {
    pmsg[l] = allocH(256 * 256);
    pu1[l]  = allocH(256 * 256);
    pu2[l]  = allocH(256 * 256);
  }
  _Float16* pco1 = allocH(256 * 128);
  _Float16* pco2 = allocH(128 * 64);
  _Float16* pct1 = allocH(256 * 256);
  _Float16* pct2 = allocH(256 * 128);
  _Float16* pnt1 = allocH(512 * 128);
  _Float16* pedt = allocH(256 * 256);
  _Float16* pedb = allocH(256 * 256);
  _Float16* pert = allocH(256 * 256);
  _Float16* perb = allocH(256 * 256);

  _Float16* z16    = allocH(64 * 128);
  _Float16* zhid16 = allocH(64 * 512);
  _Float16* Acat16 = allocH(4096 * 512);
  _Float16* x16    = allocH(4096 * 256);
  _Float16* t1h    = allocH(4096 * 256);
  _Float16* t2h    = allocH(4096 * 256);
  _Float16* mh     = allocH(4096 * 256);
  _Float16* g16    = allocH(4096 * 256);

  float* zh     = allocF(64 * 256);
  float* x      = allocF(4096 * 256);
  float* t2     = allocF(4096 * 256);
  float* g      = allocF(4096 * 256);
  float* Pd     = allocF(4096 * 256);
  float* Qd     = allocF(4096 * 256);
  float* Pr     = allocF(4096 * 256);
  float* Qr     = allocF(4096 * 256);
  float* pooled = allocF(64 * 256);
  float* fhid   = allocF(64 * 64);

  auto PACK = [&](const float* W, _Float16* P, int K, int N) {
    pack_b_kernel<<<(K * N + 255) / 256, 256, 0, stream>>>(W, P, K, N);
  };
  auto GEMM = [&](const _Float16* A, const _Float16* Bp, const float* bias,
                  float* C32, _Float16* C16, int M, int K, int N, int relu) {
    int blocks = (((M >> 4) + 7) / 8) * (N >> 5);
    gemm_wmma_kernel<<<blocks, 256, 0, stream>>>(A, Bp, bias, C32, C16, M, K, N, relu);
  };
  auto CVT = [&](const float* s, _Float16* d, int n) {
    cvt_f16_kernel<<<(n + 255) / 256, 256, 0, stream>>>(s, d, n);
  };

  // ---- one-time weight repack (f32 -> f16 WMMA fragment order) ----
  PACK(F(3), pz1, 128, 512);
  PACK(F(5), pz2, 512, 256);
  PACK(F(7), pc1, 512, 256);
  PACK(F(9), pc2, 256, 256);
  for (int l = 0; l < 4; ++l) {
    int bi = 11 + 7 * l;
    PACK(F(bi + 0), pmsg[l], 256, 256);
    PACK(F(bi + 1), pu1[l], 256, 256);
    PACK(F(bi + 3), pu2[l], 256, 256);
  }
  PACK(F(39), pco1, 256, 128);
  PACK(F(41), pco2, 128, 64);
  PACK(F(45), pct1, 256, 256);
  PACK(F(47), pct2, 256, 128);
  PACK(F(51), pnt1, 512, 128);
  PACK(F(59),             pedt, 256, 256);   // edge_dist W1 top half
  PACK(F(59) + 256 * 256, pedb, 256, 256);   // edge_dist W1 bottom half
  PACK(F(63),             pert, 256, 256);   // edge_dir  W1 top half
  PACK(F(63) + 256 * 256, perb, 256, 256);   // edge_dir  W1 bottom half

  // ---- z projection ----
  CVT(z, z16, 64 * 128);
  GEMM(z16, pz1, F(4), nullptr, zhid16, 64, 128, 512, 1);
  GEMM(zhid16, pz2, F(6), zh, nullptr, 64, 512, 256, 0);

  // ---- combine node queries with per-scene context ----
  concat_qz_kernel<<<(4096 * 512) / 256, 256, 0, stream>>>(Acat16, nq, zh);
  GEMM(Acat16, pc1, F(8), nullptr, t1h, 4096, 512, 256, 1);
  GEMM(t1h, pc2, F(10), x, x16, 4096, 256, 256, 0);

  // ---- message passing: scatter-first (linearity), then tiny GEMMs ----
  for (int l = 0; l < 4; ++l) {
    int bi = 11 + 7 * l;
    fill_zero_kernel<<<(NTOT * 256) / 256, 256, 0, stream>>>(g, NTOT * 256);
    scatter_kernel<<<(E + 7) / 8, 256, 0, stream>>>(g, x, src, dst, E);
    CVT(g, g16, NTOT * 256);
    GEMM(g16, pmsg[l], nullptr, nullptr, mh, 4096, 256, 256, 0);
    GEMM(mh, pu1[l], F(bi + 2), nullptr, t1h, 4096, 256, 256, 1);
    GEMM(t1h, pu2[l], F(bi + 4), t2, nullptr, 4096, 256, 256, 0);
    resln_kernel<<<NTOT / 8, 256, 0, stream>>>(x, x16, t2, F(bi + 5), F(bi + 6), NTOT);
  }

  // ---- node heads ----
  GEMM(x16, pco1, F(40), nullptr, t1h, 4096, 256, 128, 1);
  GEMM(t1h, pco2, F(42), nullptr, t2h, 4096, 128, 64, 1);
  head_small_kernel<<<(4096 * 3 + 255) / 256, 256, 0, stream>>>(
      t2h, F(43), F(44), out + O_CO, 4096, 64, 3, 0);

  GEMM(x16, pct1, F(46), nullptr, t1h, 4096, 256, 256, 1);
  GEMM(t1h, pct2, F(48), nullptr, t2h, 4096, 256, 128, 1);
  head_small_kernel<<<(4096 * 3 + 255) / 256, 256, 0, stream>>>(
      t2h, F(49), F(50), out + O_CT, 4096, 128, 3, 0);

  concat_xz_kernel<<<(4096 * 512) / 256, 256, 0, stream>>>(Acat16, x, zh);
  GEMM(Acat16, pnt1, F(52), nullptr, t1h, 4096, 512, 128, 1);
  head_small_kernel<<<(4096 * 3 + 255) / 256, 256, 0, stream>>>(
      t1h, F(53), F(54), out + O_NT, 4096, 128, 3, 0);

  pool_kernel<<<(64 * 256) / 256, 256, 0, stream>>>(x, pooled);
  freq1_kernel<<<(64 * 64) / 256, 256, 0, stream>>>(pooled, F(55), F(56), fhid);
  freq2_kernel<<<1, 64, 0, stream>>>(fhid, F(57), F(58), out + O_FREQ);

  // ---- edge heads: factored node tables, then wave-per-edge gather ----
  GEMM(x16, pedt, nullptr, Pd, nullptr, 4096, 256, 256, 0);
  GEMM(x16, pedb, nullptr, Qd, nullptr, 4096, 256, 256, 0);
  GEMM(x16, pert, nullptr, Pr, nullptr, 4096, 256, 256, 0);
  GEMM(x16, perb, nullptr, Qr, nullptr, 4096, 256, 256, 0);
  edge_head_kernel<<<(E + 7) / 8, 256, 0, stream>>>(
      Pd, Qd, Pr, Qr, F(60), F(61), F(62), F(64), F(65), F(66),
      src, dst, out + O_ED, out + O_ER, E);
  (void)n_in; (void)out_size; (void)ws_size;
}
